// Vitransformer_80479097193199
// MI455X (gfx1250) — matrix operand order repaired and checked by reading, not compiled, not run on hardware
//
#include <hip/hip_runtime.h>
#include <hip/hip_bf16.h>
#include <stdint.h>

// ---------------- problem constants ----------------
#define BATCH   8
#define NTOK    1024          // tokens per image (32x32 patches)
#define MTOT    8192          // BATCH * NTOK
#define DIMC    588           // embedding dim = 3*14*14
#define K1P     608           // 588 padded to 19*32
#define KT1     19            // K tiles of 32
#define NH      14
#define HD      42
#define HDP     64            // head dim padded to 2*32
#define N1TILES 40            // gemm1 col tiles (>=ceil(588/16)=37, mult of 4)
#define N2TILES 112           // gemm2 col tiles (>=ceil(1764/16)=111, mult of 4)

typedef __bf16 bf16t;
typedef __attribute__((ext_vector_type(16))) __bf16 v16bf;
typedef __attribute__((ext_vector_type(8)))  __bf16 v8bf;
typedef __attribute__((ext_vector_type(8)))  float  v8f;

#define WMMA_BF16(a, b, c) \
  __builtin_amdgcn_wmma_f32_16x16x32_bf16(false, (a), false, (b), (short)0, (c), false, false)

// Build a v16bf fragment from two contiguous 8-element (16B) chunks.
__device__ __forceinline__ v16bf ld_frag(const bf16t* p0, const bf16t* p1) {
  v8bf lo = *(const v8bf*)p0;
  v8bf hi = *(const v8bf*)p1;
  v16bf r;
#pragma unroll
  for (int i = 0; i < 8; ++i) { r[i] = lo[i]; r[i + 8] = hi[i]; }
  return r;
}

// CDNA5 async global->LDS copy (ASYNCcnt-tracked), 16 bytes.
__device__ __forceinline__ void async_copy_b128(void* lds_ptr, const void* gptr) {
  uint32_t loff = (uint32_t)(uintptr_t)lds_ptr;  // low 32 bits = LDS byte offset
  asm volatile("global_load_async_to_lds_b128 %0, %1, off"
               :: "v"(loff), "v"(gptr) : "memory");
}
__device__ __forceinline__ void wait_async() {
  asm volatile("s_wait_asynccnt 0" ::: "memory");
}

// ---------------- pack kernels ----------------

// im2col of x into bf16 A panel [MTOT, K1P]; zero pad k >= 588.
__global__ void pack_x_kernel(const float* __restrict__ x, bf16t* __restrict__ Apack) {
  int tid = blockIdx.x * 256 + threadIdx.x;
  if (tid >= MTOT * K1P) return;
  int m = tid / K1P, kk = tid % K1P;
  float v = 0.f;
  if (kk < DIMC) {
    int b = m >> 10, t = m & 1023, py = t >> 5, px = t & 31;
    int c = kk / 196, rem = kk % 196, r = rem / 14, cc = rem % 14;
    v = x[(((size_t)(b * 3 + c) * 448) + (py * 14 + r)) * 448 + (px * 14 + cc)];
  }
  Apack[(size_t)m * K1P + kk] = (bf16t)v;
}

// Pack a torch-layout [Nreal, Kreal] weight into B-fragment-native bf16 tiles:
// tile (nt,kt) holds 32 lanes x 16 contiguous bf16; lane L covers column
// n = nt*16 + (L&15), values j=0..15 -> k = kt*32 + (L>=16 ? 16 : 0) + j.
__global__ void pack_w_kernel(const float* __restrict__ W, bf16t* __restrict__ Bp,
                              int Nreal, int Kreal, int Ntiles) {
  int tid = blockIdx.x * 256 + threadIdx.x;
  if (tid >= Ntiles * KT1 * 512) return;
  int tile = tid >> 9, r = tid & 511;
  int L = r >> 4, j = r & 15;
  int nt = tile / KT1, kt = tile % KT1;
  int n  = nt * 16 + (L & 15);
  int kk = kt * 32 + ((L >> 4) * 16) + j;
  float v = (n < Nreal && kk < Kreal) ? W[(size_t)n * Kreal + kk] : 0.f;
  Bp[(size_t)tile * 512 + L * 16 + j] = (bf16t)v;
}

// ---------------- generic WMMA GEMM ----------------
// Block = 64 rows x 64 cols, 4 waves (each 16 rows x 4 col tiles). K = 608.
// Ping-pong LDS double buffering: async stage of tile kt+1 overlaps the
// 4 WMMAs + B-fragment loads of tile kt; asynccnt/barrier only at loop tail.
__global__ __launch_bounds__(128) void gemm_kernel(
    const bf16t* __restrict__ A, const bf16t* __restrict__ Bp,
    const float* __restrict__ bias, int mode, int Nreal,
    bf16t* __restrict__ tokens, bf16t* __restrict__ qb,
    bf16t* __restrict__ kb, bf16t* __restrict__ vTb) {
  __shared__ __attribute__((aligned(16))) bf16t sA[2][64 * 40];  // 32 k + 8 pad
  const int m0 = blockIdx.x * 64;
  const int ntbase = blockIdx.y * 4;
  const int wave = threadIdx.x >> 5;
  const int lane = threadIdx.x & 31;
  const int mrow = wave * 16 + (lane & 15);
  const int abase = (lane < 16) ? 0 : 16;

  auto stage = [&](int kt, int buf) {
#pragma unroll
    for (int i = 0; i < 2; ++i) {
      int chunk = threadIdx.x + i * 128;   // 256 16B chunks = 64 rows x 4
      int row = chunk >> 2, c = chunk & 3;
      async_copy_b128(&sA[buf][row * 40 + c * 8],
                      A + (size_t)(m0 + row) * K1P + kt * 32 + c * 8);
    }
  };

  v8f zero = {0.f, 0.f, 0.f, 0.f, 0.f, 0.f, 0.f, 0.f};
  v8f acc[4];
#pragma unroll
  for (int t = 0; t < 4; ++t) acc[t] = zero;

  stage(0, 0);
  wait_async();
  __syncthreads();

  for (int kt = 0; kt < KT1; ++kt) {
    const int cur = kt & 1;
    if (kt + 1 < KT1) stage(kt + 1, cur ^ 1);   // overlap DMA with compute
    // A fragment: lane row = lane&15, K chunks at {abase, abase+8}
    v16bf av = ld_frag(&sA[cur][mrow * 40 + abase],
                       &sA[cur][mrow * 40 + abase + 8]);
    v16bf bv[4];
#pragma unroll
    for (int t = 0; t < 4; ++t) {
      const bf16t* bt = Bp + ((size_t)(ntbase + t) * KT1 + kt) * 512 + lane * 16;
      bv[t] = ld_frag(bt, bt + 8);
    }
#pragma unroll
    for (int t = 0; t < 4; ++t) acc[t] = WMMA_BF16(av, bv[t], acc[t]);
    if (kt + 1 < KT1) {
      wait_async();       // own async writes to buffer cur^1 done
      __syncthreads();    // everyone's writes done / reads of cur done
    }
  }

  // epilogue: C layout -> lane col = lane&15, rows = (lane>>4)*8 + v
#pragma unroll
  for (int t = 0; t < 4; ++t) {
    int n = (ntbase + t) * 16 + (lane & 15);
    if (n >= Nreal) continue;
    float bv = bias[n];
#pragma unroll
    for (int v = 0; v < 8; ++v) {
      int m = m0 + wave * 16 + ((lane >> 4) * 8) + v;
      float val = acc[t][v] + bv;
      if (mode == 0) {
        tokens[(size_t)m * K1P + n] = (bf16t)val;   // pad cols stay zero (memset)
      } else {
        int s = n / DIMC, c = n % DIMC, h = c / HD, d = c % HD;
        int b = m >> 10, tok = m & 1023;
        size_t bh = (size_t)b * NH + h;
        if (s == 0)      qb[(bh * NTOK + tok) * HDP + d] = (bf16t)(val * 0.125f);
        else if (s == 1) kb[(bh * NTOK + tok) * HDP + d] = (bf16t)val;
        else             vTb[(bh * HDP + d) * NTOK + tok] = (bf16t)val;  // transposed
      }
    }
  }
}

// ---------------- flash attention ----------------
// Block = one (b,h) and 64 query rows; 4 waves x 16 rows. Keys looped in 64s.
__global__ __launch_bounds__(128) void attn_kernel(
    const bf16t* __restrict__ q, const bf16t* __restrict__ k,
    const bf16t* __restrict__ vT, float* __restrict__ out) {
  __shared__ __attribute__((aligned(16))) bf16t sP[4][16 * 72];
  const int bid = blockIdx.x;
  const int bh = bid >> 4;          // b*NH + h
  const int mblk = bid & 15;
  const int wave = threadIdx.x >> 5;
  const int lane = threadIdx.x & 31;
  const int m0w = mblk * 64 + wave * 16;
  const int abase = (lane < 16) ? 0 : 16;

  // q fragments (A layout), head_dim padded to 64 = 2 K-tiles
  const bf16t* qrow = q + ((size_t)bh * NTOK + m0w + (lane & 15)) * HDP;
  v16bf qa[2];
#pragma unroll
  for (int kt = 0; kt < 2; ++kt)
    qa[kt] = ld_frag(qrow + kt * 32 + abase, qrow + kt * 32 + abase + 8);

  v8f zero = {0.f, 0.f, 0.f, 0.f, 0.f, 0.f, 0.f, 0.f};
  v8f oacc[3];
#pragma unroll
  for (int f = 0; f < 3; ++f) oacc[f] = zero;
  float mrow[8], lrow[8];
#pragma unroll
  for (int v = 0; v < 8; ++v) { mrow[v] = -3.0e38f; lrow[v] = 0.f; }

  bf16t* myP = sP[wave];

  for (int kbb = 0; kbb < NTOK; kbb += 64) {
    // S = q @ k^T for 4 key tiles; preload both K-chunks per tile
    v8f sfr[4];
#pragma unroll
    for (int t = 0; t < 4; ++t) {
      const bf16t* krow = k + ((size_t)bh * NTOK + kbb + t * 16 + (lane & 15)) * HDP;
      v16bf kv[2];
#pragma unroll
      for (int kt = 0; kt < 2; ++kt) {
        const bf16t* kp = krow + kt * 32 + ((lane >> 4) * 16);
        kv[kt] = ld_frag(kp, kp + 8);
      }
      v8f st = zero;
      st = WMMA_BF16(qa[0], kv[0], st);
      st = WMMA_BF16(qa[1], kv[1], st);
      sfr[t] = st;
    }
    // online softmax; each row lives in one 16-lane half (shfl_xor 1/2/4/8)
#pragma unroll
    for (int v = 0; v < 8; ++v) {
      float bm = fmaxf(fmaxf(sfr[0][v], sfr[1][v]), fmaxf(sfr[2][v], sfr[3][v]));
#pragma unroll
      for (int msk = 1; msk < 16; msk <<= 1) bm = fmaxf(bm, __shfl_xor(bm, msk, 32));
      float mnew = fmaxf(mrow[v], bm);
      float al = __expf(mrow[v] - mnew);
      float rs = 0.f;
#pragma unroll
      for (int t = 0; t < 4; ++t) {
        float p = __expf(sfr[t][v] - mnew);
        sfr[t][v] = p;
        rs += p;
      }
#pragma unroll
      for (int msk = 1; msk < 16; msk <<= 1) rs += __shfl_xor(rs, msk, 32);
      lrow[v] = lrow[v] * al + rs;
      mrow[v] = mnew;
#pragma unroll
      for (int f = 0; f < 3; ++f) oacc[f][v] *= al;
    }
    // restage P into A layout via per-wave LDS tile (same-wave DS ordering)
#pragma unroll
    for (int t = 0; t < 4; ++t)
#pragma unroll
      for (int v = 0; v < 8; ++v)
        myP[((lane >> 4) * 8 + v) * 72 + t * 16 + (lane & 15)] = (bf16t)sfr[t][v];
    // O += P @ V  (V^T rows are features -> contiguous B fragments)
#pragma unroll
    for (int kt = 0; kt < 2; ++kt) {
      const bf16t* pp = &myP[(lane & 15) * 72 + kt * 32 + abase];
      v16bf pa = ld_frag(pp, pp + 8);
      v16bf vv[3];
#pragma unroll
      for (int f = 0; f < 3; ++f) {
        const bf16t* vp = vT + ((size_t)bh * HDP + f * 16 + (lane & 15)) * NTOK
                        + kbb + kt * 32 + ((lane >> 4) * 16);
        vv[f] = ld_frag(vp, vp + 8);
      }
#pragma unroll
      for (int f = 0; f < 3; ++f) oacc[f] = WMMA_BF16(pa, vv[f], oacc[f]);
    }
  }

  const int b = bh / NH, h = bh % NH;
#pragma unroll
  for (int f = 0; f < 3; ++f) {
    int d = f * 16 + (lane & 15);
    if (d >= HD) continue;
#pragma unroll
    for (int v = 0; v < 8; ++v) {
      int m = m0w + (lane >> 4) * 8 + v;
      out[((size_t)(b * NTOK + m)) * DIMC + h * HD + d] = oacc[f][v] / lrow[v];
    }
  }
}

// ---------------- host launcher ----------------
static inline size_t align256(size_t x) { return (x + 255) & ~(size_t)255; }

extern "C" void kernel_launch(void* const* d_in, const int* in_sizes, int n_in,
                              void* d_out, int out_size, void* d_ws, size_t ws_size,
                              hipStream_t stream) {
  (void)in_sizes; (void)n_in; (void)out_size; (void)ws_size;
  const float* x      = (const float*)d_in[0];
  const float* conv_w = (const float*)d_in[1];
  const float* conv_b = (const float*)d_in[2];
  const float* qkv_w  = (const float*)d_in[3];
  const float* qkv_b  = (const float*)d_in[4];
  float* out = (float*)d_out;

  char* ws = (char*)d_ws;
  size_t off = 0;
  bf16t* Apack  = (bf16t*)(ws + off); off += align256((size_t)MTOT * K1P * 2);
  bf16t* tokens = (bf16t*)(ws + off); off += align256((size_t)MTOT * K1P * 2);
  bf16t* Wc     = (bf16t*)(ws + off); off += align256((size_t)N1TILES * KT1 * 512 * 2);
  bf16t* Wq     = (bf16t*)(ws + off); off += align256((size_t)N2TILES * KT1 * 512 * 2);
  bf16t* qb     = (bf16t*)(ws + off); off += align256((size_t)BATCH * NH * NTOK * HDP * 2);
  bf16t* kb     = (bf16t*)(ws + off); off += align256((size_t)BATCH * NH * NTOK * HDP * 2);
  bf16t* vTb    = (bf16t*)(ws + off); off += align256((size_t)BATCH * NH * HDP * NTOK * 2);

  // zero staging (pad columns / padded head-dim rows must read as 0)
  hipMemsetAsync(d_ws, 0, off, stream);

  { int n = MTOT * K1P;
    pack_x_kernel<<<(n + 255) / 256, 256, 0, stream>>>(x, Apack); }
  { int n = N1TILES * KT1 * 512;
    pack_w_kernel<<<(n + 255) / 256, 256, 0, stream>>>(conv_w, Wc, DIMC, DIMC, N1TILES); }
  { int n = N2TILES * KT1 * 512;
    pack_w_kernel<<<(n + 255) / 256, 256, 0, stream>>>(qkv_w, Wq, 3 * DIMC, DIMC, N2TILES); }

  gemm_kernel<<<dim3(MTOT / 64, N1TILES / 4), 128, 0, stream>>>(
      Apack, Wc, conv_b, 0, DIMC, tokens, nullptr, nullptr, nullptr);
  gemm_kernel<<<dim3(MTOT / 64, N2TILES / 4), 128, 0, stream>>>(
      tokens, Wq, qkv_b, 1, 3 * DIMC, nullptr, qb, kb, vTb);

  attn_kernel<<<BATCH * NH * (NTOK / 64), 128, 0, stream>>>(qb, kb, vTb, out);
}
